// EncoderLayer_49374944035459
// MI455X (gfx1250) — compile-verified, matrix operand order checked by reading, not run on hardware
//
#include <hip/hip_runtime.h>
#include <hip/hip_bf16.h>
#include <cstdint>
#include <cstddef>

// ---- problem constants (match reference) ----
#define TB   4
#define TS   1024
#define TD   1024
#define THH  16
#define TDK  64
#define TDFF 4096
#define TM   (TB * TS)          // 4096 rows of activations

// ---- WMMA vector types ----
typedef __attribute__((ext_vector_type(16))) __bf16 v16bf;
typedef __attribute__((ext_vector_type(8)))  __bf16 v8bf;
typedef __attribute__((ext_vector_type(8)))  float  v8f;

__device__ __forceinline__ unsigned short f2bf(float f) {
    union { float f; unsigned u; } cv; cv.f = f;
    unsigned u = cv.u;
    return (unsigned short)((u + 0x7FFFu + ((u >> 16) & 1u)) >> 16);
}

// Build a 16-element bf16 fragment from two contiguous 8-element (16B) chunks.
__device__ __forceinline__ v16bf load_frag(const unsigned short* p0,
                                           const unsigned short* p1) {
    v8bf lo = *(const v8bf*)p0;
    v8bf hi = *(const v8bf*)p1;
    return __builtin_shufflevector(lo, hi, 0,1,2,3,4,5,6,7,8,9,10,11,12,13,14,15);
}

__device__ __forceinline__ v8f wmma_bf16(v16bf a, v16bf b, v8f c) {
    return __builtin_amdgcn_wmma_f32_16x16x32_bf16(false, a, false, b,
                                                   (short)0, c, false, false);
}

// =====================================================================
// fp32 -> bf16 conversion (vectorized)
// =====================================================================
__global__ __launch_bounds__(256) void cvt_bf16_kernel(const float* __restrict__ src,
                                                       unsigned short* __restrict__ dst,
                                                       int n) {
    int i = (blockIdx.x * 256 + threadIdx.x) * 4;
    if (i >= n) return;
    float4 v = *(const float4*)(src + i);
    unsigned long long pack =
        (unsigned long long)f2bf(v.x)         |
        ((unsigned long long)f2bf(v.y) << 16) |
        ((unsigned long long)f2bf(v.z) << 32) |
        ((unsigned long long)f2bf(v.w) << 48);
    *(unsigned long long*)(dst + i) = pack;
}

// =====================================================================
// Tiled bf16 WMMA GEMM:  C[M,N] = A[M,K] @ B[K,N] + bias, optional relu.
// 256 threads = 8 waves, 128x128 tile, BK=32.
// Each wave computes a 64x32 block = 4x2 grid of 16x16 WMMA tiles.
// =====================================================================
#define BM 128
#define BN 128
#define BK 32

__global__ __launch_bounds__(256) void gemm_bf16_kernel(
    const unsigned short* __restrict__ A,   // [M,K] bf16 row-major
    const unsigned short* __restrict__ Bw,  // [K,N] bf16 row-major
    const float* __restrict__ bias,         // [N]
    float* __restrict__ Cf,                 // fp32 out (or null)
    unsigned short* __restrict__ Cb,        // bf16 out (or null)
    int M, int N, int K, int relu)
{
    __shared__ __align__(16) unsigned short Asub[BM][BK];
    __shared__ __align__(16) unsigned short Bsub[BN][BK];  // B stored transposed

    const int tid  = threadIdx.x;
    const int wid  = tid >> 5;
    const int lane = tid & 31;
    const int m0   = blockIdx.y * BM;
    const int n0   = blockIdx.x * BN;
    const int wm   = (wid & 1) * 64;
    const int wn   = (wid >> 1) * 32;
    const int lrow  = lane & 15;
    const int lhalf = lane >> 4;
    const int k0l   = lhalf * 8;

    v8f zero = {};
    v8f acc[4][2];
    #pragma unroll
    for (int i = 0; i < 4; ++i)
        #pragma unroll
        for (int j = 0; j < 2; ++j) acc[i][j] = zero;

    const int arow = tid >> 1;            // A-stage: row 0..127
    const int acol = (tid & 1) * 16;      // cols {0,16}
    const int bk   = tid >> 3;            // B-stage: k row 0..31
    const int bn   = (tid & 7) * 16;      // n cols {0..112}

    for (int kk = 0; kk < K; kk += BK) {
        // ---- stage A tile (128 x 32), 16 ushorts per thread ----
        {
            const uint4* g = (const uint4*)(A + (size_t)(m0 + arow) * K + kk + acol);
            uint4 v0 = g[0];
            uint4 v1 = g[1];
            *(uint4*)&Asub[arow][acol]     = v0;
            *(uint4*)&Asub[arow][acol + 8] = v1;
        }
        // ---- stage B tile (32 x 128) transposed into Bsub[n][k] ----
        {
            const uint4* g = (const uint4*)(Bw + (size_t)(kk + bk) * N + n0 + bn);
            uint4 v0 = g[0];
            uint4 v1 = g[1];
            unsigned short tmp[16];
            *(uint4*)&tmp[0] = v0;
            *(uint4*)&tmp[8] = v1;
            #pragma unroll
            for (int i = 0; i < 16; ++i) Bsub[bn + i][bk] = tmp[i];
        }
        __syncthreads();

        // prefetch next K-tile (global_prefetch_b8)
        if (kk + BK < K) {
            __builtin_prefetch(A + (size_t)(m0 + arow) * K + kk + BK + acol, 0, 1);
            __builtin_prefetch(Bw + (size_t)(kk + BK + bk) * N + n0 + bn, 0, 1);
        }

        v16bf af[4], bf[2];
        #pragma unroll
        for (int i = 0; i < 4; ++i) {
            int r = wm + i * 16 + lrow;
            af[i] = load_frag(&Asub[r][k0l], &Asub[r][k0l + 16]);
        }
        #pragma unroll
        for (int j = 0; j < 2; ++j) {
            int c = wn + j * 16 + lrow;
            bf[j] = load_frag(&Bsub[c][k0l], &Bsub[c][k0l + 16]);
        }
        #pragma unroll
        for (int i = 0; i < 4; ++i)
            #pragma unroll
            for (int j = 0; j < 2; ++j)
                acc[i][j] = wmma_bf16(af[i], bf[j], acc[i][j]);
        __syncthreads();
    }

    // ---- epilogue: bias (+relu), store fp32 and/or bf16 ----
    #pragma unroll
    for (int j = 0; j < 2; ++j) {
        int col = n0 + wn + j * 16 + lrow;
        float bv = bias ? bias[col] : 0.f;
        #pragma unroll
        for (int i = 0; i < 4; ++i) {
            #pragma unroll
            for (int r = 0; r < 8; ++r) {
                int row = m0 + wm + i * 16 + r + lhalf * 8;
                float val = acc[i][j][r] + bv;
                if (relu) val = fmaxf(val, 0.f);
                if (Cf) Cf[(size_t)row * N + col] = val;
                if (Cb) Cb[(size_t)row * N + col] = f2bf(val);
            }
        }
    }
}

// =====================================================================
// Flash attention with relative-position bias.
// Grid: B*H*(S/64) blocks, 128 threads (4 waves). Wave = 16 query rows.
// Streams keys in steps of 32; online softmax; P re-laid out via LDS.
// =====================================================================
__global__ __launch_bounds__(128) void attn_kernel(
    const unsigned short* __restrict__ Q,    // [B,S,D] bf16 (head-concat)
    const unsigned short* __restrict__ Km,   // [B,S,D] bf16
    const unsigned short* __restrict__ Vm,   // [B,S,D] bf16
    const float* __restrict__ rel_table,     // [2*S-1, H]
    const int*   __restrict__ mask,          // [B,1,1,S]
    unsigned short* __restrict__ Ctx)        // [B,S,D] bf16
{
    __shared__ __align__(16) unsigned short Vt[TDK][32];       // V tile transposed
    __shared__ __align__(16) unsigned short Plds[4][16][32];   // per-wave P scratch

    const int tid  = threadIdx.x;
    const int wid  = tid >> 5;
    const int lane = tid & 31;
    const int qt = blockIdx.x & 15;          // S/64 = 16 query tiles
    const int h  = (blockIdx.x >> 4) & 15;
    const int b  = blockIdx.x >> 8;
    const int qb = qt * 64 + wid * 16;
    const int lrow  = lane & 15;
    const int lhalf = lane >> 4;
    const int k0l   = lhalf * 8;

    // ---- Q fragments (16 rows x 64 dims -> two 16x32 A fragments) ----
    v16bf qf[2];
    {
        const unsigned short* p = Q + ((size_t)b * TS + qb + lrow) * TD + h * TDK;
        qf[0] = load_frag(p + k0l,      p + k0l + 16);
        qf[1] = load_frag(p + 32 + k0l, p + 32 + k0l + 16);
    }

    v8f zero = {};
    v8f o[4];
    #pragma unroll
    for (int i = 0; i < 4; ++i) o[i] = zero;
    float mrun[8], lrun[8];
    #pragma unroll
    for (int r = 0; r < 8; ++r) { mrun[r] = -1e30f; lrun[r] = 0.f; }

    const int vkey = tid >> 2;               // V-stage: key 0..31
    const int vd0  = (tid & 3) * 16;         // d chunk {0,16,32,48}

    for (int kt = 0; kt < TS; kt += 32) {
        // ---- stage V tile (32 keys x 64 dims) transposed into Vt[d][key] ----
        {
            const unsigned short* vp =
                Vm + ((size_t)b * TS + kt + vkey) * TD + h * TDK + vd0;
            uint4 v0 = *(const uint4*)vp;
            uint4 v1 = *(const uint4*)(vp + 8);
            unsigned short tmp[16];
            *(uint4*)&tmp[0] = v0;
            *(uint4*)&tmp[8] = v1;
            #pragma unroll
            for (int i = 0; i < 16; ++i) Vt[vd0 + i][vkey] = tmp[i];
        }
        __syncthreads();

        // ---- scores: two 16x16 tiles (keys kt..kt+31) ----
        v8f s[2];
        #pragma unroll
        for (int nt = 0; nt < 2; ++nt) {
            s[nt] = zero;
            int key = kt + nt * 16 + lrow;   // this lane's B-operand column
            const unsigned short* kp = Km + ((size_t)b * TS + key) * TD + h * TDK;
            v16bf kf0 = load_frag(kp + k0l,      kp + k0l + 16);
            v16bf kf1 = load_frag(kp + 32 + k0l, kp + 32 + k0l + 16);
            s[nt] = wmma_bf16(qf[0], kf0, s[nt]);
            s[nt] = wmma_bf16(qf[1], kf1, s[nt]);
        }

        // ---- scale + rel-pos bias + mask; per-row max ----
        float rowmax[8];
        #pragma unroll
        for (int r = 0; r < 8; ++r) rowmax[r] = -1e30f;
        #pragma unroll
        for (int nt = 0; nt < 2; ++nt) {
            int k = kt + nt * 16 + lrow;
            int mk = mask[b * TS + k];
            #pragma unroll
            for (int r = 0; r < 8; ++r) {
                int q = qb + r + lhalf * 8;
                int c = q - k + (TS - 1);
                c = c < 0 ? 0 : (c > 2 * TS - 2 ? 2 * TS - 2 : c);
                float sv = s[nt][r] * 0.125f + rel_table[c * THH + h];
                if (mk == 0) sv = -1e9f;
                s[nt][r] = sv;
                rowmax[r] = fmaxf(rowmax[r], sv);
            }
        }
        // reduce max across the 16 lanes holding each row
        #pragma unroll
        for (int off = 1; off < 16; off <<= 1)
            #pragma unroll
            for (int r = 0; r < 8; ++r)
                rowmax[r] = fmaxf(rowmax[r], __shfl_xor(rowmax[r], off, 32));

        // ---- online softmax update ----
        float scale[8], psum[8];
        #pragma unroll
        for (int r = 0; r < 8; ++r) {
            float nm = fmaxf(mrun[r], rowmax[r]);
            scale[r] = __expf(mrun[r] - nm);
            mrun[r] = nm;
            psum[r] = 0.f;
        }
        #pragma unroll
        for (int nt = 0; nt < 2; ++nt)
            #pragma unroll
            for (int r = 0; r < 8; ++r) {
                float p = __expf(s[nt][r] - mrun[r]);
                s[nt][r] = p;
                psum[r] += p;
            }
        #pragma unroll
        for (int off = 1; off < 16; off <<= 1)
            #pragma unroll
            for (int r = 0; r < 8; ++r)
                psum[r] += __shfl_xor(psum[r], off, 32);
        #pragma unroll
        for (int r = 0; r < 8; ++r) lrun[r] = lrun[r] * scale[r] + psum[r];
        #pragma unroll
        for (int nd = 0; nd < 4; ++nd)
            #pragma unroll
            for (int r = 0; r < 8; ++r) o[nd][r] *= scale[r];

        // ---- P (C-layout) -> bf16 A-fragment via per-wave LDS round trip ----
        #pragma unroll
        for (int nt = 0; nt < 2; ++nt)
            #pragma unroll
            for (int r = 0; r < 8; ++r)
                Plds[wid][r + lhalf * 8][nt * 16 + lrow] = f2bf(s[nt][r]);
        // LDS ops within a wave are in-order: store->load is safe here
        v16bf pa;
        {
            const unsigned short* pp = &Plds[wid][lrow][0];
            pa = load_frag(pp + k0l, pp + k0l + 16);
        }

        // ---- O += P @ V  (four 16x16 output tiles over the 64-dim head) ----
        #pragma unroll
        for (int nd = 0; nd < 4; ++nd) {
            const unsigned short* vp = &Vt[nd * 16 + lrow][0];
            v16bf vb = load_frag(vp + k0l, vp + k0l + 16);
            o[nd] = wmma_bf16(pa, vb, o[nd]);
        }
        __syncthreads();   // protect Vt before restage
    }

    // ---- normalize and store ctx (bf16, head-concatenated) ----
    #pragma unroll
    for (int r = 0; r < 8; ++r) {
        float inv = 1.f / lrun[r];
        int q = qb + r + lhalf * 8;
        #pragma unroll
        for (int nd = 0; nd < 4; ++nd) {
            int d = h * TDK + nd * 16 + lrow;
            Ctx[((size_t)b * TS + q) * TD + d] = f2bf(o[nd][r] * inv);
        }
    }
}

// =====================================================================
// Fused residual + LayerNorm. One block (256 threads) per row of D=1024.
// Optionally emits both fp32 and bf16 outputs.
// =====================================================================
__global__ __launch_bounds__(256) void ln_kernel(
    const float* __restrict__ X, const float* __restrict__ Res,
    const float* __restrict__ g, const float* __restrict__ beta,
    float* __restrict__ outf, unsigned short* __restrict__ outb)
{
    __shared__ float red[256];
    const int row = blockIdx.x;
    const int tid = threadIdx.x;
    const float* xr = X   + (size_t)row * TD;
    const float* rr = Res + (size_t)row * TD;

    float vals[4];
    float sum = 0.f;
    #pragma unroll
    for (int i = 0; i < 4; ++i) {
        int c = tid + i * 256;
        vals[i] = xr[c] + rr[c];
        sum += vals[i];
    }
    red[tid] = sum;
    __syncthreads();
    for (int s = 128; s > 0; s >>= 1) {
        if (tid < s) red[tid] += red[tid + s];
        __syncthreads();
    }
    float mean = red[0] * (1.f / TD);
    __syncthreads();

    float vs = 0.f;
    #pragma unroll
    for (int i = 0; i < 4; ++i) {
        float d = vals[i] - mean;
        vs += d * d;
    }
    red[tid] = vs;
    __syncthreads();
    for (int s = 128; s > 0; s >>= 1) {
        if (tid < s) red[tid] += red[tid + s];
        __syncthreads();
    }
    float rstd = rsqrtf(red[0] * (1.f / TD) + 1e-5f);

    #pragma unroll
    for (int i = 0; i < 4; ++i) {
        int c = tid + i * 256;
        float y = (vals[i] - mean) * rstd * g[c] + beta[c];
        if (outf) outf[(size_t)row * TD + c] = y;
        if (outb) outb[(size_t)row * TD + c] = f2bf(y);
    }
}

// =====================================================================
// host launcher
// =====================================================================
extern "C" void kernel_launch(void* const* d_in, const int* in_sizes, int n_in,
                              void* d_out, int out_size, void* d_ws, size_t ws_size,
                              hipStream_t stream) {
    (void)in_sizes; (void)n_in; (void)out_size; (void)ws_size;
    const float* x   = (const float*)d_in[0];
    const float* wq  = (const float*)d_in[1];
    const float* bq  = (const float*)d_in[2];
    const float* wk  = (const float*)d_in[3];
    const float* bk  = (const float*)d_in[4];
    const float* wv  = (const float*)d_in[5];
    const float* bv  = (const float*)d_in[6];
    const float* wo  = (const float*)d_in[7];
    const float* bo  = (const float*)d_in[8];
    const float* rel = (const float*)d_in[9];
    const float* g1  = (const float*)d_in[10];
    const float* be1 = (const float*)d_in[11];
    const float* w1  = (const float*)d_in[12];
    const float* b1  = (const float*)d_in[13];
    const float* w2  = (const float*)d_in[14];
    const float* b2  = (const float*)d_in[15];
    const float* g2  = (const float*)d_in[16];
    const float* be2 = (const float*)d_in[17];
    const int*   msk = (const int*)d_in[18];

    char* ws = (char*)d_ws;
    const size_t MB = 1u << 20;
    unsigned short* xb   = (unsigned short*)(ws + 0 * MB);    // 8 MB
    unsigned short* wqb  = (unsigned short*)(ws + 8 * MB);    // 2 MB
    unsigned short* wkb  = (unsigned short*)(ws + 10 * MB);   // 2 MB
    unsigned short* wvb  = (unsigned short*)(ws + 12 * MB);   // 2 MB
    unsigned short* wob  = (unsigned short*)(ws + 14 * MB);   // 2 MB
    unsigned short* w1b  = (unsigned short*)(ws + 16 * MB);   // 8 MB
    unsigned short* w2b  = (unsigned short*)(ws + 24 * MB);   // 8 MB
    unsigned short* Qb   = (unsigned short*)(ws + 32 * MB);   // 8 MB
    unsigned short* Kb   = (unsigned short*)(ws + 40 * MB);   // 8 MB
    unsigned short* Vb   = (unsigned short*)(ws + 48 * MB);   // 8 MB
    unsigned short* ctxb = (unsigned short*)(ws + 56 * MB);   // 8 MB
    float* attnproj      = (float*)(ws + 64 * MB);            // 16 MB
    float* x1f           = (float*)(ws + 80 * MB);            // 16 MB
    unsigned short* x1b  = (unsigned short*)(ws + 96 * MB);   // 8 MB
    unsigned short* ffh  = (unsigned short*)(ws + 104 * MB);  // 32 MB
    float* ff2           = (float*)(ws + 136 * MB);           // 16 MB

    auto cvt = [&](const float* s, unsigned short* d, int n) {
        cvt_bf16_kernel<<<(n / 4 + 255) / 256, 256, 0, stream>>>(s, d, n);
    };
    cvt(x,  xb,  TM * TD);
    cvt(wq, wqb, TD * TD);
    cvt(wk, wkb, TD * TD);
    cvt(wv, wvb, TD * TD);
    cvt(wo, wob, TD * TD);
    cvt(w1, w1b, TD * TDFF);
    cvt(w2, w2b, TDFF * TD);

    // QKV projections  (M=4096, N=1024, K=1024) -> bf16
    dim3 gD(TD / BN, TM / BM);
    gemm_bf16_kernel<<<gD, 256, 0, stream>>>(xb, wqb, bq, nullptr, Qb,
                                             TM, TD, TD, 0);
    gemm_bf16_kernel<<<gD, 256, 0, stream>>>(xb, wkb, bk, nullptr, Kb,
                                             TM, TD, TD, 0);
    gemm_bf16_kernel<<<gD, 256, 0, stream>>>(xb, wvb, bv, nullptr, Vb,
                                             TM, TD, TD, 0);

    // flash attention: B*H*(S/64) = 1024 blocks of 128 threads
    attn_kernel<<<TB * THH * (TS / 64), 128, 0, stream>>>(Qb, Kb, Vb, rel, msk, ctxb);

    // output projection -> fp32 attnproj
    gemm_bf16_kernel<<<gD, 256, 0, stream>>>(ctxb, wob, bo, attnproj, nullptr,
                                             TM, TD, TD, 0);

    // x1 = LN(x + attnproj)  (fp32 + bf16 copies)
    ln_kernel<<<TM, 256, 0, stream>>>(x, attnproj, g1, be1, x1f, x1b);

    // FFN1: relu(x1 @ w1 + b1) -> bf16   (M=4096, N=4096, K=1024)
    dim3 gF1(TDFF / BN, TM / BM);
    gemm_bf16_kernel<<<gF1, 256, 0, stream>>>(x1b, w1b, b1, nullptr, ffh,
                                              TM, TDFF, TD, 1);

    // FFN2: ffh @ w2 + b2 -> fp32        (M=4096, N=1024, K=4096)
    gemm_bf16_kernel<<<gD, 256, 0, stream>>>(ffh, w2b, b2, ff2, nullptr,
                                             TM, TD, TDFF, 0);

    // out = LN(x1 + ff2)
    ln_kernel<<<TM, 256, 0, stream>>>(x1f, ff2, g2, be2, (float*)d_out, nullptr);
}
